// HofmanSwarm_57028575756550
// MI455X (gfx1250) — matrix operand order repaired and checked by reading, not compiled, not run on hardware
//
#include <hip/hip_runtime.h>
#include <hip/hip_bf16.h>
#include <math.h>

#define NB 16      // batch
#define NT 128     // time steps
#define ND 256     // model dim
#define NA 64      // agents
#define NSEN 16    // sensory agents
#define NV 32000   // vocab

typedef __bf16 bf16;
typedef __attribute__((ext_vector_type(16))) __bf16 v16bf;
typedef __attribute__((ext_vector_type(8)))  float  v8f;

// ---------------------------------------------------------------- fragments
// A: 16x32 bf16 (MxK), row-major source. Layout per ISA 7.12.2:
// lanes 0-15: M=lane, VGPR0-3 hold K=0..7, VGPR4-7 hold K=16..23
// lanes 16-31: M=lane-16, VGPR0-3 hold K=8..15, VGPR4-7 hold K=24..31
__device__ __forceinline__ v16bf load_a(const bf16* __restrict__ A, int lda) {
  int lane = threadIdx.x & 31;
  const bf16* rp = A + (size_t)(lane & 15) * lda + ((lane >> 4) << 3);
  v16bf a;
#pragma unroll
  for (int j = 0; j < 8; ++j) {
    int k = ((j & 4) << 2) + ((j & 3) << 1); // 0,2,4,6,16,18,20,22
    a[2 * j]     = rp[k];
    a[2 * j + 1] = rp[k + 1];
  }
  return a;
}

// B: 32x16 bf16 (KxN), loaded from transposed storage BT[N][K] so each lane's
// K-values are contiguous. lanes 0-15: N=lane, K=0..7,16..23; lanes 16-31: K=8..15,24..31.
__device__ __forceinline__ v16bf load_bt(const bf16* __restrict__ BT, int ldb) {
  int lane = threadIdx.x & 31;
  const bf16* cp = BT + (size_t)(lane & 15) * ldb + ((lane >> 4) << 3);
  v16bf b;
#pragma unroll
  for (int j = 0; j < 16; ++j) b[j] = cp[j + (j & 8)];
  return b;
}

__device__ __forceinline__ v8f wmma_bf16(v16bf a, v16bf b, v8f c) {
  return __builtin_amdgcn_wmma_f32_16x16x32_bf16(false, a, false, b, (short)0, c,
                                                 false, false);
}

__device__ __forceinline__ float gelu_exact(float x) {
  return 0.5f * x * (1.0f + erff(x * 0.70710678118654752f));
}
__device__ __forceinline__ float sigmoidf(float x) {
  return 1.0f / (1.0f + __expf(-x));
}

// ---------------------------------------------------------------- setup kernels
__global__ void k_embed(const int* __restrict__ idx, const float* __restrict__ perc,
                        const float* __restrict__ pos, bf16* __restrict__ out) {
  int i = blockIdx.x * blockDim.x + threadIdx.x; // B*T*D
  int d = i & (ND - 1);
  int bt = i >> 8;           // b*T + t
  int t = bt & (NT - 1);
  out[i] = (bf16)(perc[(size_t)idx[bt] * ND + d] + pos[t * ND + d]);
}

// out[n*K + k] = bf16(in[k*N + n])  (weight -> transposed bf16 for B operand)
__global__ void k_transpose_bf(const float* __restrict__ in, bf16* __restrict__ out,
                               int K, int N) {
  long i = (long)blockIdx.x * blockDim.x + threadIdx.x;
  if (i >= (long)K * N) return;
  int n = (int)(i % N);
  long k = i / N;
  out[(size_t)n * K + k] = (bf16)in[i];
}

__global__ void k_cvt_bf(const float* __restrict__ in, bf16* __restrict__ out, int n) {
  int i = blockIdx.x * blockDim.x + threadIdx.x;
  if (i < n) out[i] = (bf16)in[i];
}

__global__ void k_zero(float* __restrict__ p, int n) {
  int i = blockIdx.x * blockDim.x + threadIdx.x;
  if (i < n) p[i] = 0.0f;
}

// softmax of agent_importance (64) and channel_mix (2), single lane
__global__ void k_iw(const float* __restrict__ ai, const float* __restrict__ cm,
                     float* __restrict__ iw, float* __restrict__ cw) {
  if (threadIdx.x != 0) return;
  float mx = -1e30f;
  for (int n = 0; n < NA; ++n) mx = fmaxf(mx, ai[n]);
  float s = 0.f;
  for (int n = 0; n < NA; ++n) s += __expf(ai[n] - mx);
  for (int n = 0; n < NA; ++n) iw[n] = __expf(ai[n] - mx) / s;
  float m2 = fmaxf(cm[0], cm[1]);
  float e0 = __expf(cm[0] - m2), e1 = __expf(cm[1] - m2);
  cw[0] = e0 / (e0 + e1);
  cw[1] = e1 / (e0 + e1);
}

// ---------------------------------------------------------------- per-step kernels
// sens = percept_t @ sensory_w + b, added into states[:, :16, :]
// A = percepts_bf[:, t, :] (M=16 batches, K=256), BT = sensT [4096 x 256]
__global__ void k_sens(const bf16* __restrict__ percepts_bf, const bf16* __restrict__ sensT,
                       const float* __restrict__ sens_b, float* __restrict__ states, int t) {
  int wave = (blockIdx.x * blockDim.x + threadIdx.x) >> 5; // 0..255 = N tile
  const bf16* A = percepts_bf + (size_t)t * ND;            // row stride T*D
  const bf16* BT = sensT + (size_t)wave * 16 * ND;
  v8f acc = {};
#pragma unroll
  for (int k0 = 0; k0 < ND; k0 += 32)
    acc = wmma_bf16(load_a(A + k0, NT * ND), load_bt(BT + k0, ND), acc);
  int lane = threadIdx.x & 31;
  int n = lane & 15, mb = (lane >> 4) * 8;
#pragma unroll
  for (int e = 0; e < 8; ++e) {
    int b = mb + e;
    int gn = wave * 16 + n;     // 0..4095
    int agent = gn >> 8, d = gn & 255;
    states[((size_t)b * NA + agent) * ND + d] += acc[e] + sens_b[gn];
  }
}

// states f32 -> bf16 (row-major) + transposed bf16 [B][D][NA]
__global__ void k_cvt_states(const float* __restrict__ states, bf16* __restrict__ s_bf,
                             bf16* __restrict__ sT_bf) {
  int i = blockIdx.x * blockDim.x + threadIdx.x; // B*NA*D
  float v = states[i];
  s_bf[i] = (bf16)v;
  int d = i & 255, n = (i >> 8) & 63, b = i >> 14;
  sT_bf[((size_t)b * ND + d) * NA + n] = (bf16)v;
}

// interaction[c][b] = W_c (64x64) @ states[b] (64x256), output bf16
__global__ void k_inter(const bf16* __restrict__ W_bf, const bf16* __restrict__ sT_bf,
                        bf16* __restrict__ inter_bf) {
  int wave = (blockIdx.x * blockDim.x + threadIdx.x) >> 5; // 0..2047
  int nt = wave & 15, mt = (wave >> 4) & 3, b = (wave >> 6) & 15, c = wave >> 10;
  const bf16* A = W_bf + c * NA * NA + mt * 16 * NA;
  const bf16* BT = sT_bf + (size_t)b * ND * NA + (size_t)nt * 16 * NA;
  v8f acc = {};
#pragma unroll
  for (int k0 = 0; k0 < NA; k0 += 32)
    acc = wmma_bf16(load_a(A + k0, NA), load_bt(BT + k0, NA), acc);
  int lane = threadIdx.x & 31;
  int n = lane & 15, mb = (lane >> 4) * 8;
  bf16* O = inter_bf + (size_t)(c * NB + b) * NA * ND;
#pragma unroll
  for (int e = 0; e < 8; ++e)
    O[(size_t)(mt * 16 + mb + e) * ND + nt * 16 + n] = (bf16)acc[e];
}

// gate_pre[c] = concat(states, inter_c) [1024x512] @ gate_w_c -> [1024x256] f32
__global__ void k_gate(const bf16* __restrict__ s_bf, const bf16* __restrict__ inter_bf,
                       const bf16* __restrict__ gate_wT, float* __restrict__ gate_pre) {
  int wave = (blockIdx.x * blockDim.x + threadIdx.x) >> 5; // 0..2047
  int nt = wave & 15, mt = (wave >> 4) & 63, c = wave >> 10;
  const bf16* As = s_bf + (size_t)mt * 16 * ND;
  const bf16* Ai = inter_bf + (size_t)c * NB * NA * ND + (size_t)mt * 16 * ND;
  const bf16* BT = gate_wT + (size_t)c * ND * 512 + (size_t)nt * 16 * 512;
  v8f acc = {};
#pragma unroll
  for (int k0 = 0; k0 < 512; k0 += 32) {
    const bf16* A = (k0 < ND) ? (As + k0) : (Ai + (k0 - ND));
    acc = wmma_bf16(load_a(A, ND), load_bt(BT + k0, 512), acc);
  }
  int lane = threadIdx.x & 31;
  int n = lane & 15, mb = (lane >> 4) * 8;
  float* O = gate_pre + (size_t)c * NB * NA * ND;
#pragma unroll
  for (int e = 0; e < 8; ++e)
    O[(size_t)(mt * 16 + mb + e) * ND + nt * 16 + n] = acc[e];
}

// cand_pre[c] = inter_c [1024x256] @ cand_w_c -> [1024x256] f32
__global__ void k_cand(const bf16* __restrict__ inter_bf, const bf16* __restrict__ cand_wT,
                       float* __restrict__ cand_pre) {
  int wave = (blockIdx.x * blockDim.x + threadIdx.x) >> 5; // 0..2047
  int nt = wave & 15, mt = (wave >> 4) & 63, c = wave >> 10;
  const bf16* A = inter_bf + (size_t)c * NB * NA * ND + (size_t)mt * 16 * ND;
  const bf16* BT = cand_wT + (size_t)c * ND * ND + (size_t)nt * 16 * ND;
  v8f acc = {};
#pragma unroll
  for (int k0 = 0; k0 < ND; k0 += 32)
    acc = wmma_bf16(load_a(A + k0, ND), load_bt(BT + k0, ND), acc);
  int lane = threadIdx.x & 31;
  int n = lane & 15, mb = (lane >> 4) * 8;
  float* O = cand_pre + (size_t)c * NB * NA * ND;
#pragma unroll
  for (int e = 0; e < 8; ++e)
    O[(size_t)(mt * 16 + mb + e) * ND + nt * 16 + n] = acc[e];
}

// gate/candidate activations, GRU-style blend, LayerNorm, channel mix.
// One wave (32 lanes x 8 elems) per row of [B*NA, D].
__global__ void k_mix(const float* __restrict__ gate_pre, const float* __restrict__ cand_pre,
                      float* __restrict__ states,
                      const float* __restrict__ gb0, const float* __restrict__ cb0,
                      const float* __restrict__ lg0, const float* __restrict__ lb0,
                      const float* __restrict__ gb1, const float* __restrict__ cb1,
                      const float* __restrict__ lg1, const float* __restrict__ lb1,
                      const float* __restrict__ cw) {
  int row = (blockIdx.x * blockDim.x + threadIdx.x) >> 5; // 0..1023
  int lane = threadIdx.x & 31;
  float* Sp = states + (size_t)row * ND;
  float s[8], out[8];
#pragma unroll
  for (int e = 0; e < 8; ++e) { s[e] = Sp[lane * 8 + e]; out[e] = 0.f; }
#pragma unroll
  for (int c = 0; c < 2; ++c) {
    const float* gp = gate_pre + (size_t)c * NB * NA * ND + (size_t)row * ND;
    const float* cp = cand_pre + (size_t)c * NB * NA * ND + (size_t)row * ND;
    const float* gb = c ? gb1 : gb0;
    const float* cb = c ? cb1 : cb0;
    const float* lg = c ? lg1 : lg0;
    const float* lb = c ? lb1 : lb0;
    float h[8], sum = 0.f, sq = 0.f;
#pragma unroll
    for (int e = 0; e < 8; ++e) {
      int d = lane * 8 + e;
      float g = sigmoidf(gp[d] + gb[d]);
      float cd = gelu_exact(cp[d] + cb[d]);
      h[e] = g * cd + (1.0f - g) * s[e];
      sum += h[e];
      sq += h[e] * h[e];
    }
#pragma unroll
    for (int m = 1; m < 32; m <<= 1) {
      sum += __shfl_xor(sum, m, 32);
      sq  += __shfl_xor(sq, m, 32);
    }
    float mu = sum * (1.0f / ND);
    float var = sq * (1.0f / ND) - mu * mu;
    float rinv = rsqrtf(var + 1e-5f);
    float cwc = cw[c];
#pragma unroll
    for (int e = 0; e < 8; ++e) {
      int d = lane * 8 + e;
      out[e] += cwc * ((h[e] - mu) * rinv * lg[d] + lb[d]);
    }
  }
#pragma unroll
  for (int e = 0; e < 8; ++e) Sp[lane * 8 + e] = out[e];
}

// consensus[b,d] = sum_n iw[n]*states[b,n,d] -> bf16
__global__ void k_consensus(const float* __restrict__ states, const float* __restrict__ iw,
                            bf16* __restrict__ cons_bf) {
  int i = blockIdx.x * blockDim.x + threadIdx.x; // B*D
  int b = i >> 8, d = i & 255;
  float acc = 0.f;
#pragma unroll 4
  for (int n = 0; n < NA; ++n)
    acc += iw[n] * states[((size_t)b * NA + n) * ND + d];
  cons_bf[i] = (bf16)acc;
}

// hidden = gelu(consensus @ act_w1 + b1), bf16 [16 x 512]
__global__ void k_head1(const bf16* __restrict__ cons_bf, const bf16* __restrict__ w1T,
                        const float* __restrict__ b1, bf16* __restrict__ hidden_bf) {
  int wave = (blockIdx.x * blockDim.x + threadIdx.x) >> 5; // 0..31 = N tile
  const bf16* BT = w1T + (size_t)wave * 16 * ND;
  v8f acc = {};
#pragma unroll
  for (int k0 = 0; k0 < ND; k0 += 32)
    acc = wmma_bf16(load_a(cons_bf + k0, ND), load_bt(BT + k0, ND), acc);
  int lane = threadIdx.x & 31;
  int n = lane & 15, mb = (lane >> 4) * 8;
#pragma unroll
  for (int e = 0; e < 8; ++e) {
    int col = wave * 16 + n;
    hidden_bf[(size_t)(mb + e) * 512 + col] = (bf16)gelu_exact(acc[e] + b1[col]);
  }
}

// logits = hidden @ act_w2 + b2 -> d_out[b, t, :]
__global__ void k_head2(const bf16* __restrict__ hidden_bf, const bf16* __restrict__ w2T,
                        const float* __restrict__ b2, float* __restrict__ out, int t) {
  int wave = (blockIdx.x * blockDim.x + threadIdx.x) >> 5; // 0..1999 = N tile
  const bf16* BT = w2T + (size_t)wave * 16 * 512;
  v8f acc = {};
#pragma unroll
  for (int k0 = 0; k0 < 512; k0 += 32)
    acc = wmma_bf16(load_a(hidden_bf + k0, 512), load_bt(BT + k0, 512), acc);
  int lane = threadIdx.x & 31;
  int n = lane & 15, mb = (lane >> 4) * 8;
#pragma unroll
  for (int e = 0; e < 8; ++e) {
    int b = mb + e;
    int v = wave * 16 + n;
    out[((size_t)b * NT + t) * NV + v] = acc[e] + b2[v];
  }
}

// ---------------------------------------------------------------- launch
extern "C" void kernel_launch(void* const* d_in, const int* in_sizes, int n_in,
                              void* d_out, int out_size, void* d_ws, size_t ws_size,
                              hipStream_t stream) {
  const int*   idx        = (const int*)  d_in[0];
  const float* perception = (const float*)d_in[1];
  const float* pos_emb    = (const float*)d_in[2];
  const float* sensory_w  = (const float*)d_in[3];
  const float* sensory_b  = (const float*)d_in[4];
  const float* channel_mix= (const float*)d_in[5];
  const float* agent_imp  = (const float*)d_in[6];
  const float* act_w1     = (const float*)d_in[7];
  const float* act_b1     = (const float*)d_in[8];
  const float* act_w2     = (const float*)d_in[9];
  const float* act_b2     = (const float*)d_in[10];
  const float* W0      = (const float*)d_in[11];
  const float* gate_w0 = (const float*)d_in[12];
  const float* gate_b0 = (const float*)d_in[13];
  const float* cand_w0 = (const float*)d_in[14];
  const float* cand_b0 = (const float*)d_in[15];
  const float* ln_g0   = (const float*)d_in[16];
  const float* ln_b0   = (const float*)d_in[17];
  const float* W1      = (const float*)d_in[18];
  const float* gate_w1 = (const float*)d_in[19];
  const float* gate_b1 = (const float*)d_in[20];
  const float* cand_w1 = (const float*)d_in[21];
  const float* cand_b1 = (const float*)d_in[22];
  const float* ln_g1   = (const float*)d_in[23];
  const float* ln_b1   = (const float*)d_in[24];

  // workspace bump allocator (256B aligned)
  char* p = (char*)d_ws;
  auto alloc = [&](size_t bytes) -> char* {
    char* r = p;
    p += (bytes + 255) & ~(size_t)255;
    return r;
  };
  bf16* percepts_bf = (bf16*)alloc((size_t)NB * NT * ND * 2);
  bf16* sensT       = (bf16*)alloc((size_t)NSEN * ND * ND * 2);      // [4096 x 256]
  bf16* gwT         = (bf16*)alloc((size_t)2 * ND * 512 * 2);        // [c][256 x 512]
  bf16* cwT         = (bf16*)alloc((size_t)2 * ND * ND * 2);         // [c][256 x 256]
  bf16* Wbf         = (bf16*)alloc((size_t)2 * NA * NA * 2);
  bf16* w1T         = (bf16*)alloc((size_t)512 * ND * 2);            // [512 x 256]
  bf16* w2T         = (bf16*)alloc((size_t)NV * 512 * 2);            // [32000 x 512]
  float* states     = (float*)alloc((size_t)NB * NA * ND * 4);
  bf16* s_bf        = (bf16*)alloc((size_t)NB * NA * ND * 2);
  bf16* sT_bf       = (bf16*)alloc((size_t)NB * ND * NA * 2);
  bf16* inter_bf    = (bf16*)alloc((size_t)2 * NB * NA * ND * 2);
  float* gate_pre   = (float*)alloc((size_t)2 * NB * NA * ND * 4);
  float* cand_pre   = (float*)alloc((size_t)2 * NB * NA * ND * 4);
  bf16* cons_bf     = (bf16*)alloc((size_t)NB * ND * 2);
  bf16* hidden_bf   = (bf16*)alloc((size_t)NB * 512 * 2);
  float* iw         = (float*)alloc(NA * 4);
  float* cwv        = (float*)alloc(2 * 4);

  // ---- setup: embed + weight conversion (bf16, B operands transposed) ----
  k_embed<<<(NB * NT * ND) / 256, 256, 0, stream>>>(idx, perception, pos_emb, percepts_bf);
  k_transpose_bf<<<(ND * NSEN * ND + 255) / 256, 256, 0, stream>>>(sensory_w, sensT, ND, NSEN * ND);
  k_transpose_bf<<<(512 * ND + 255) / 256, 256, 0, stream>>>(gate_w0, gwT, 512, ND);
  k_transpose_bf<<<(512 * ND + 255) / 256, 256, 0, stream>>>(gate_w1, gwT + (size_t)ND * 512, 512, ND);
  k_transpose_bf<<<(ND * ND + 255) / 256, 256, 0, stream>>>(cand_w0, cwT, ND, ND);
  k_transpose_bf<<<(ND * ND + 255) / 256, 256, 0, stream>>>(cand_w1, cwT + (size_t)ND * ND, ND, ND);
  k_cvt_bf<<<(NA * NA + 255) / 256, 256, 0, stream>>>(W0, Wbf, NA * NA);
  k_cvt_bf<<<(NA * NA + 255) / 256, 256, 0, stream>>>(W1, Wbf + NA * NA, NA * NA);
  k_transpose_bf<<<(ND * 512 + 255) / 256, 256, 0, stream>>>(act_w1, w1T, ND, 512);
  k_transpose_bf<<<((size_t)512 * NV + 255) / 256, 256, 0, stream>>>(act_w2, w2T, 512, NV);
  k_zero<<<(NB * NA * ND) / 256, 256, 0, stream>>>(states, NB * NA * ND);
  k_iw<<<1, 32, 0, stream>>>(agent_imp, channel_mix, iw, cwv);

  // ---- sequential scan over T ----
  for (int t = 0; t < NT; ++t) {
    k_sens<<<64, 128, 0, stream>>>(percepts_bf, sensT, sensory_b, states, t);
    k_cvt_states<<<(NB * NA * ND) / 256, 256, 0, stream>>>(states, s_bf, sT_bf);
    k_inter<<<512, 128, 0, stream>>>(Wbf, sT_bf, inter_bf);
    k_gate<<<512, 128, 0, stream>>>(s_bf, inter_bf, gwT, gate_pre);
    k_cand<<<512, 128, 0, stream>>>(inter_bf, cwT, cand_pre);
    k_mix<<<128, 256, 0, stream>>>(gate_pre, cand_pre, states,
                                   gate_b0, cand_b0, ln_g0, ln_b0,
                                   gate_b1, cand_b1, ln_g1, ln_b1, cwv);
    k_consensus<<<(NB * ND) / 256, 256, 0, stream>>>(states, iw, cons_bf);
    k_head1<<<8, 128, 0, stream>>>(cons_bf, w1T, act_b1, hidden_bf);
    k_head2<<<500, 128, 0, stream>>>(hidden_bf, w2T, act_b2, (float*)d_out, t);
  }
}
